// SRGNNLayer_52055003627520
// MI455X (gfx1250) — matrix operand order repaired and checked by reading, not compiled, not run on hardware
//
#include <hip/hip_runtime.h>
#include <stdint.h>

#define DD 128
#define TD 384   // 3*D

typedef __attribute__((ext_vector_type(16))) __bf16         v16bf;
typedef __attribute__((ext_vector_type(8)))  float          v8f;
typedef __attribute__((ext_vector_type(8)))  unsigned short u16x8;
typedef __attribute__((ext_vector_type(16))) unsigned short u16x16;

__device__ __forceinline__ unsigned short f32_to_bf16_rne(float f) {
    union { float f; unsigned int u; } v; v.f = f;
    unsigned int u = v.u;
    unsigned int r = u + 0x7FFFu + ((u >> 16) & 1u);
    return (unsigned short)(r >> 16);
}

// ---------------- kernel 1: zero the scatter accumulators ----------------
__global__ void zero_f4(float4* p, size_t n4) {
    size_t i = (size_t)blockIdx.x * blockDim.x + threadIdx.x;
    size_t stride = (size_t)gridDim.x * blockDim.x;
    float4 z = make_float4(0.f, 0.f, 0.f, 0.f);
    for (; i < n4; i += stride) p[i] = z;
}

// ---------------- kernel 2: fold W1/W2 into W_ih, convert to bf16 --------
// M1 = W_ih[:, :D] @ W1   (gi = neigh1 @ M1^T + neigh2 @ M2^T + b_ih)
// M2 = W_ih[:, D:] @ W2
// Whh = bf16(W_hh)
__global__ __launch_bounds__(128) void prep_weights(
    const float* __restrict__ W1, const float* __restrict__ W2,
    const float* __restrict__ W_ih, const float* __restrict__ W_hh,
    unsigned short* __restrict__ M1, unsigned short* __restrict__ M2,
    unsigned short* __restrict__ Whh)
{
    int r = blockIdx.x;    // 0..383
    int c = threadIdx.x;   // 0..127
    float a1 = 0.f, a2 = 0.f;
    #pragma unroll 4
    for (int j = 0; j < DD; ++j) {
        float wij0 = W_ih[(size_t)r * (2 * DD) + j];
        float wij1 = W_ih[(size_t)r * (2 * DD) + DD + j];
        a1 += wij0 * W1[(size_t)j * DD + c];
        a2 += wij1 * W2[(size_t)j * DD + c];
    }
    M1[(size_t)r * DD + c]  = f32_to_bf16_rne(a1);
    M2[(size_t)r * DD + c]  = f32_to_bf16_rne(a2);
    Whh[(size_t)r * DD + c] = f32_to_bf16_rne(W_hh[(size_t)r * DD + c]);
}

// ---------------- kernel 3: bidirectional weighted scatter ----------------
// one wave per edge; lane handles 4 consecutive features (32*4 = 128)
__global__ __launch_bounds__(256) void edge_scatter(
    const float* __restrict__ feat,
    const long long* __restrict__ esrc, const long long* __restrict__ edst,
    const float* __restrict__ ew,
    float* __restrict__ sm1, float* __restrict__ sw1,
    float* __restrict__ sm2, float* __restrict__ sw2, int E)
{
    int e = blockIdx.x * 8 + (threadIdx.x >> 5);
    if (e >= E) return;
    int lane = threadIdx.x & 31;
    long long s = esrc[e], d = edst[e];
    float w = ew[e];
    float4 fs = ((const float4*)(feat + (size_t)s * DD))[lane];
    float4 fd = ((const float4*)(feat + (size_t)d * DD))[lane];
    float* o1 = sm1 + (size_t)d * DD + lane * 4;
    float* o2 = sm2 + (size_t)s * DD + lane * 4;
    atomicAdd(o1 + 0, fs.x * w); atomicAdd(o1 + 1, fs.y * w);
    atomicAdd(o1 + 2, fs.z * w); atomicAdd(o1 + 3, fs.w * w);
    atomicAdd(o2 + 0, fd.x * w); atomicAdd(o2 + 1, fd.y * w);
    atomicAdd(o2 + 2, fd.z * w); atomicAdd(o2 + 3, fd.w * w);
    if (lane == 0) { atomicAdd(sw1 + d, w); atomicAdd(sw2 + s, w); }
}

// ---------------- WMMA fragment helpers (CDNA5 bf16 layouts) --------------
// A (16x32 bf16): lane half h, row m=lane&15; elems 0..7 -> K = kb*32 + 8h + e,
//                 elems 8..15 -> K = kb*32 + 16 + 8h + (e-8)  (two 16B chunks)
__device__ __forceinline__ v16bf load_a_frag(const unsigned short* rowp, int half, int kbase) {
    u16x8 c0 = *(const u16x8*)(rowp + kbase + 8 * half);
    u16x8 c1 = *(const u16x8*)(rowp + kbase + 16 + 8 * half);
    u16x16 cc;
    #pragma unroll
    for (int i = 0; i < 8; ++i) { cc[i] = c0[i]; cc[i + 8] = c1[i]; }
    return __builtin_bit_cast(v16bf, cc);
}
// B (32x16 bf16): lane half h, col n=lane&15; elems 0..7 -> K = kb*32 + 8h + e,
// elems 8..15 -> K = kb*32 + 16 + 8h + (e-8); B[k][n] = M[row = rt*16+n][k]  (two 16B chunks)
__device__ __forceinline__ v16bf load_b_frag(const unsigned short* M, int row, int half, int kbase) {
    const unsigned short* rp = M + (size_t)row * DD + kbase;
    u16x8 c0 = *(const u16x8*)(rp + 8 * half);
    u16x8 c1 = *(const u16x8*)(rp + 16 + 8 * half);
    u16x16 v;
    #pragma unroll
    for (int i = 0; i < 8; ++i) { v[i] = c0[i]; v[i + 8] = c1[i]; }
    return __builtin_bit_cast(v16bf, v);
}

// ---------------- kernel 4: fused normalize + GEMM (WMMA) + GRU ----------
__global__ __launch_bounds__(128) void fused_gemm_gru(
    const float* __restrict__ feat,
    const float* __restrict__ sm1, const float* __restrict__ sw1,
    const float* __restrict__ sm2, const float* __restrict__ sw2,
    const unsigned short* __restrict__ M1, const unsigned short* __restrict__ M2,
    const unsigned short* __restrict__ Whh,
    const float* __restrict__ b_ih, const float* __restrict__ b_hh,
    float* __restrict__ out, int N)
{
    __shared__ __align__(16) unsigned short A1[16][DD];   // bf16 neigh1 tile
    __shared__ __align__(16) unsigned short A2[16][DD];   // bf16 neigh2 tile
    __shared__ __align__(16) unsigned short AF[16][DD];   // bf16 feat tile
    __shared__ __align__(16) float GI[16][TD];            // gi accum (f32)
    __shared__ __align__(16) float GH[16][TD];            // gh accum (f32)

    const int tid = threadIdx.x;
    const int node0 = blockIdx.x * 16;

    // Phase 0: normalize weighted sums, convert tiles to bf16 in LDS
    for (int i = tid; i < 16 * DD; i += 128) {
        int m = i >> 7, c = i & (DD - 1);
        int node = node0 + m;
        float v1 = 0.f, v2 = 0.f, vf = 0.f;
        if (node < N) {
            float s1 = sw1[node], s2 = sw2[node];
            v1 = (s1 > 0.f) ? sm1[(size_t)node * DD + c] / s1 : 0.f;
            v2 = (s2 > 0.f) ? sm2[(size_t)node * DD + c] / s2 : 0.f;
            vf = feat[(size_t)node * DD + c];
        }
        A1[m][c] = f32_to_bf16_rne(v1);
        A2[m][c] = f32_to_bf16_rne(v2);
        AF[m][c] = f32_to_bf16_rne(vf);
    }
    __syncthreads();

    const int wave = tid >> 5;
    const int lane = tid & 31;
    const int half = lane >> 4;
    const int mrow = lane & 15;   // A row for this lane / C column index n

    // Hoist ALL A-fragments into registers once (3 matrices x 4 k-blocks).
    // 96 VGPRs; removes ds loads + dscnt waits from the WMMA inner loop.
    v16bf a1f[4], a2f[4], aff[4];
    #pragma unroll
    for (int kb = 0; kb < 4; ++kb) {
        a1f[kb] = load_a_frag(&A1[mrow][0], half, kb * 32);
        a2f[kb] = load_a_frag(&A2[mrow][0], half, kb * 32);
        aff[kb] = load_a_frag(&AF[mrow][0], half, kb * 32);
    }

    // Phase 1: 48 output tiles (24 gi, 24 gh), 4 waves round-robin.
    for (int job = wave; job < 48; job += 4) {
        // prefetch next job's B row (L2-resident; global_prefetch_b8)
        int nj = job + 4;
        if (nj < 48) {
            const unsigned short* nB = (nj < 24) ? M1 : Whh;
            int nrow = ((nj < 24) ? nj : nj - 24) * 16 + mrow;
            __builtin_prefetch(nB + (size_t)nrow * DD, 0, 1);
        }

        v8f acc = {};
        if (job < 24) {                       // gi tile rt: neigh1*M1 + neigh2*M2
            int rt = job;
            int brow = rt * 16 + mrow;
            #pragma unroll
            for (int kb = 0; kb < 4; ++kb) {
                v16bf b = load_b_frag(M1, brow, half, kb * 32);
                acc = __builtin_amdgcn_wmma_f32_16x16x32_bf16(false, a1f[kb], false, b,
                                                              (short)0, acc, false, false);
            }
            #pragma unroll
            for (int kb = 0; kb < 4; ++kb) {
                v16bf b = load_b_frag(M2, brow, half, kb * 32);
                acc = __builtin_amdgcn_wmma_f32_16x16x32_bf16(false, a2f[kb], false, b,
                                                              (short)0, acc, false, false);
            }
            float bias = b_ih[brow];
            #pragma unroll
            for (int g = 0; g < 8; ++g)       // C: vgpr g -> M=g+8*half, N=lane&15
                GI[g + 8 * half][rt * 16 + mrow] = acc[g] + bias;
        } else {                              // gh tile rt: feat*Whh
            int rt = job - 24;
            int brow = rt * 16 + mrow;
            #pragma unroll
            for (int kb = 0; kb < 4; ++kb) {
                v16bf b = load_b_frag(Whh, brow, half, kb * 32);
                acc = __builtin_amdgcn_wmma_f32_16x16x32_bf16(false, aff[kb], false, b,
                                                              (short)0, acc, false, false);
            }
            float bias = b_hh[brow];
            #pragma unroll
            for (int g = 0; g < 8; ++g)
                GH[g + 8 * half][rt * 16 + mrow] = acc[g] + bias;
        }
    }
    __syncthreads();

    // Phase 2: GRU gates in f32
    for (int i = tid; i < 16 * DD; i += 128) {
        int m = i >> 7, c = i & (DD - 1);
        int node = node0 + m;
        if (node >= N) continue;
        float ir = GI[m][c], iz = GI[m][c + DD], inn = GI[m][c + 2 * DD];
        float hr = GH[m][c], hz = GH[m][c + DD], hnn = GH[m][c + 2 * DD];
        float r  = 1.f / (1.f + __expf(-(ir + hr)));
        float z  = 1.f / (1.f + __expf(-(iz + hz)));
        float ng = tanhf(inn + r * hnn);
        float h  = feat[(size_t)node * DD + c];
        out[(size_t)node * DD + c] = (1.f - z) * ng + z * h;
    }
}

// -------------------------------- launcher --------------------------------
extern "C" void kernel_launch(void* const* d_in, const int* in_sizes, int n_in,
                              void* d_out, int out_size, void* d_ws, size_t ws_size,
                              hipStream_t stream) {
    const float*     feat = (const float*)d_in[0];
    const long long* esrc = (const long long*)d_in[1];
    const long long* edst = (const long long*)d_in[2];
    const float*     ew   = (const float*)d_in[3];
    const float*     W1   = (const float*)d_in[4];
    const float*     W2   = (const float*)d_in[5];
    const float*     W_ih = (const float*)d_in[6];
    const float*     W_hh = (const float*)d_in[7];
    const float*     b_ih = (const float*)d_in[8];
    const float*     b_hh = (const float*)d_in[9];
    float* out = (float*)d_out;

    const int N = in_sizes[0] / DD;
    const int E = in_sizes[1];

    // workspace layout
    float* sm1 = (float*)d_ws;
    float* sm2 = sm1 + (size_t)N * DD;
    float* sw1 = sm2 + (size_t)N * DD;
    float* sw2 = sw1 + N;
    uintptr_t p = ((uintptr_t)(sw2 + N) + 31) & ~(uintptr_t)31;
    unsigned short* M1  = (unsigned short*)p;
    unsigned short* M2  = M1 + (size_t)TD * DD;
    unsigned short* Whh = M2 + (size_t)TD * DD;

    // 1) zero accumulators (2*N*D + 2*N floats, divisible by 4)
    size_t n4 = ((size_t)2 * N * DD + 2 * (size_t)N) / 4;
    int zgrid = (int)((n4 + 255) / 256);
    zero_f4<<<zgrid, 256, 0, stream>>>((float4*)sm1, n4);

    // 2) fold weights, convert to bf16
    prep_weights<<<TD, 128, 0, stream>>>(W1, W2, W_ih, W_hh, M1, M2, Whh);

    // 3) bidirectional weighted scatter (one wave per edge)
    edge_scatter<<<(E + 7) / 8, 256, 0, stream>>>(feat, esrc, edst, ew,
                                                  sm1, sw1, sm2, sw2, E);

    // 4) fused normalize + WMMA GEMMs + GRU gates (one 16-node tile per block)
    fused_gemm_gru<<<(N + 15) / 16, 128, 0, stream>>>(feat, sm1, sw1, sm2, sw2,
                                                      M1, M2, Whh, b_ih, b_hh, out, N);
}
